// Attention_65481071394960
// MI455X (gfx1250) — matrix operand order repaired and checked by reading, not compiled, not run on hardware
//
#include <hip/hip_runtime.h>
#include <hip/hip_bf16.h>
#include <math.h>

typedef __attribute__((ext_vector_type(16))) __bf16 v16bf;
typedef __attribute__((ext_vector_type(8)))  __bf16 v8bf;
typedef __attribute__((ext_vector_type(4)))  __bf16 v4bf;
typedef __attribute__((ext_vector_type(8)))  float  v8f;

#define S_LEN  4096
#define H_DIM  1024
#define NHEAD  16
#define HDIM   64
#define G_DIM  4096   /* 4*H */

// ------------------------------------------------------------ f32 -> bf16

__global__ __launch_bounds__(256) void cvt_f32_bf16(const float* __restrict__ src,
                                                    __bf16* __restrict__ dst) {
  const size_t i = (size_t)blockIdx.x * 256 + threadIdx.x;
  const float4 f = ((const float4*)src)[i];
  v4bf o;
  o[0] = (__bf16)f.x; o[1] = (__bf16)f.y; o[2] = (__bf16)f.z; o[3] = (__bf16)f.w;
  ((v4bf*)dst)[i] = o;
}

// ------------------------------------------------------------ fragments

// A fragment: 16x32 bf16 (MxK); lane L -> row M = L%16;
// elems 0..7 = K kb+0..7 (kb = 0|8 by half-wave), elems 8..15 = 16+kb+0..7.
static __device__ __forceinline__ v16bf load_fragA(const __bf16* base, int lda,
                                                   int m0, int k0, int lane) {
  const int m  = m0 + (lane & 15);
  const int kb = (lane & 16) ? 8 : 0;
  const __bf16* row = base + (size_t)m * lda + k0 + kb;
  const v8bf lo = *(const v8bf*)row;
  const v8bf hi = *(const v8bf*)(row + 16);
  v16bf u;
#pragma unroll
  for (int e = 0; e < 8; ++e) { u[e] = lo[e]; u[8 + e] = hi[e]; }
  return u;
}

// Same fragment from an fp32 source (single-use operands): native cvt.
static __device__ __forceinline__ v16bf load_fragA(const float* base, int lda,
                                                   int m0, int k0, int lane) {
  const int m  = m0 + (lane & 15);
  const int kb = (lane & 16) ? 8 : 0;
  const float* row = base + (size_t)m * lda + k0 + kb;
  v16bf u;
#pragma unroll
  for (int e = 0; e < 8; ++e) { u[e] = (__bf16)row[e]; u[8 + e] = (__bf16)row[16 + e]; }
  return u;
}

// B fragment: 32x16 bf16 (KxN) from row-major [n, k] (implicit transpose).
// Lane L -> col N = L%16; elems 0..7 = K kb+0..7 (kb = 0|8 by half-wave), elems 8..15 = 16+kb+0..7.
static __device__ __forceinline__ v16bf load_fragB(const __bf16* base, int ldb,
                                                   int n0, int k0, int lane) {
  const int n  = n0 + (lane & 15);
  const int kb = (lane & 16) ? 8 : 0;
  const __bf16* row = base + (size_t)n * ldb + k0 + kb;
  const v8bf lo = *(const v8bf*)row;
  const v8bf hi = *(const v8bf*)(row + 16);
  v16bf u;
#pragma unroll
  for (int e = 0; e < 8; ++e) { u[e] = lo[e]; u[8 + e] = hi[e]; }
  return u;
}

// ------------------------------------------------------------ WMMA GEMM
// C[z][m][n] = scale * sum_k A[z][m][k] * B[z][n][k]  (+bias0[n] +bias1[n])
// causal!=0: keep n<=m else -inf. Block tile 128x64, 8 waves, 4 accs/wave.
// k-loop unrolled x2 with ping-pong fragment buffers (no rotation copies);
// requires K % 64 == 0 (holds for all uses: 64, 1024, 4096).
template <typename TA>
__global__ __launch_bounds__(256) void gemm_wmma(
    const TA* __restrict__ A, long long sA, int lda,
    const __bf16* __restrict__ B, long long sB, int ldb,
    float* __restrict__ C, long long sC, int ldc,
    const float* __restrict__ bias0, const float* __restrict__ bias1,
    int M, int N, int K, float scale, int causal) {
  const int lane = threadIdx.x & 31;
  const int wave = threadIdx.x >> 5;
  const int m0   = blockIdx.y * 128 + wave * 16;
  const int n0   = blockIdx.x * 64;
  const int z    = blockIdx.z;

  A += (long long)z * sA;
  B += (long long)z * sB;
  C += (long long)z * sC;

  const int mr = ((lane >> 4) & 1) * 8;
  const int nc = lane & 15;

  if (causal && n0 > blockIdx.y * 128 + 127) {
#pragma unroll
    for (int nb = 0; nb < 4; ++nb)
#pragma unroll
      for (int r = 0; r < 8; ++r)
        C[(size_t)(m0 + mr + r) * ldc + (n0 + nb * 16 + nc)] = -INFINITY;
    return;
  }

  v8f acc[4] = {v8f{0}, v8f{0}, v8f{0}, v8f{0}};

  // ---- ping buffers: k-step 0
  v16bf aP = load_fragA(A, lda, m0, 0, lane);
  v16bf bP[4];
#pragma unroll
  for (int nb = 0; nb < 4; ++nb) bP[nb] = load_fragB(B, ldb, n0 + nb * 16, 0, lane);

  int k0 = 0;
  for (; k0 + 64 < K; k0 += 64) {
    // pong = k0+32, issued before consuming ping
    v16bf aQ = load_fragA(A, lda, m0, k0 + 32, lane);
    v16bf bQ[4];
#pragma unroll
    for (int nb = 0; nb < 4; ++nb) bQ[nb] = load_fragB(B, ldb, n0 + nb * 16, k0 + 32, lane);

#pragma unroll
    for (int nb = 0; nb < 4; ++nb)
      acc[nb] = __builtin_amdgcn_wmma_f32_16x16x32_bf16(
          false, aP, false, bP[nb], (short)0, acc[nb], false, false);

    // ping = k0+64, issued before consuming pong
    aP = load_fragA(A, lda, m0, k0 + 64, lane);
#pragma unroll
    for (int nb = 0; nb < 4; ++nb) bP[nb] = load_fragB(B, ldb, n0 + nb * 16, k0 + 64, lane);

#pragma unroll
    for (int nb = 0; nb < 4; ++nb)
      acc[nb] = __builtin_amdgcn_wmma_f32_16x16x32_bf16(
          false, aQ, false, bQ[nb], (short)0, acc[nb], false, false);
  }

  // ---- drain: steps K-64 (in ping) and K-32
  {
    v16bf aQ = load_fragA(A, lda, m0, K - 32, lane);
    v16bf bQ[4];
#pragma unroll
    for (int nb = 0; nb < 4; ++nb) bQ[nb] = load_fragB(B, ldb, n0 + nb * 16, K - 32, lane);

#pragma unroll
    for (int nb = 0; nb < 4; ++nb)
      acc[nb] = __builtin_amdgcn_wmma_f32_16x16x32_bf16(
          false, aP, false, bP[nb], (short)0, acc[nb], false, false);
#pragma unroll
    for (int nb = 0; nb < 4; ++nb)
      acc[nb] = __builtin_amdgcn_wmma_f32_16x16x32_bf16(
          false, aQ, false, bQ[nb], (short)0, acc[nb], false, false);
  }

#pragma unroll
  for (int nb = 0; nb < 4; ++nb) {
#pragma unroll
    for (int r = 0; r < 8; ++r) {
      const int m = m0 + mr + r;
      const int n = n0 + nb * 16 + nc;
      float v = acc[nb][r];
      if (bias0) v += bias0[n];
      if (bias1) v += bias1[n];
      if (causal) v = (n <= m) ? v * scale : -INFINITY;
      else if (scale != 1.0f) v *= scale;
      C[(size_t)m * ldc + n] = v;
    }
  }
}

// ------------------------------------------------------------ LSTM scan

static __device__ __forceinline__ void grid_barrier(unsigned* cnt, unsigned* gen,
                                                    unsigned nb) {
  __syncthreads();
  if (threadIdx.x == 0) {
    unsigned g = __hip_atomic_load(gen, __ATOMIC_ACQUIRE, __HIP_MEMORY_SCOPE_AGENT);
    if (__hip_atomic_fetch_add(cnt, 1u, __ATOMIC_ACQ_REL, __HIP_MEMORY_SCOPE_AGENT)
        == nb - 1) {
      __hip_atomic_store(cnt, 0u, __ATOMIC_RELAXED, __HIP_MEMORY_SCOPE_AGENT);
      __hip_atomic_fetch_add(gen, 1u, __ATOMIC_ACQ_REL, __HIP_MEMORY_SCOPE_AGENT);
    } else {
      while (__hip_atomic_load(gen, __ATOMIC_ACQUIRE, __HIP_MEMORY_SCOPE_AGENT) == g)
        __builtin_amdgcn_s_sleep(2);
    }
  }
  __syncthreads();
}

#define LSTM_NB 32
__global__ __launch_bounds__(256) void lstm_scan(
    const float* __restrict__ xproj,  // [S, 4096]
    const float* __restrict__ Whh,    // [4096, 1024]
    float* __restrict__ hs,           // [S, 1024] out
    float* __restrict__ hbuf, float* __restrict__ cbuf,
    unsigned* __restrict__ bar) {
  __shared__ float sh[H_DIM];
  __shared__ float sg[128];
  const int tid = threadIdx.x;
  const int blk = blockIdx.x;

  if (tid < 32) { hbuf[blk * 32 + tid] = 0.0f; cbuf[blk * 32 + tid] = 0.0f; }
  grid_barrier(bar, bar + 1, LSTM_NB);

  for (int t = 0; t < S_LEN; ++t) {
    for (int i = tid; i < H_DIM; i += 256) sh[i] = hbuf[i];
    __syncthreads();

    if (tid < 128) {
      const int g = tid >> 5;
      const int j = tid & 31;
      const int row = g * H_DIM + blk * 32 + j;
      const float* w = Whh + (size_t)row * H_DIM;
      float acc = xproj[(size_t)t * G_DIM + row];
#pragma unroll 8
      for (int k = 0; k < H_DIM; ++k) acc = fmaf(sh[k], w[k], acc);
      sg[tid] = acc;
    }
    __syncthreads();

    if (tid < 32) {
      const int j = tid;
      const float iv = 1.0f / (1.0f + __expf(-sg[j]));
      const float fv = 1.0f / (1.0f + __expf(-sg[32 + j]));
      const float gv = tanhf(sg[64 + j]);
      const float ov = 1.0f / (1.0f + __expf(-sg[96 + j]));
      const float c  = fv * cbuf[blk * 32 + j] + iv * gv;
      const float h  = ov * tanhf(c);
      cbuf[blk * 32 + j] = c;
      hbuf[blk * 32 + j] = h;
      hs[(size_t)t * H_DIM + blk * 32 + j] = h;
    }
    grid_barrier(bar, bar + 1, LSTM_NB);
  }
}

// ------------------------------------------------------------ small ops

__global__ __launch_bounds__(256) void rope_kernel(
    float* __restrict__ q, float* __restrict__ k,
    const float* __restrict__ cosT, const float* __restrict__ sinT) {
  const size_t idx = (size_t)blockIdx.x * 256 + threadIdx.x;  // S*NH*32
  const int s    = (int)(idx >> 9);
  const int r    = (int)(idx & 511);
  const int head = r >> 5;
  const int d    = r & 31;
  float* x = (blockIdx.y == 0) ? q : k;
  const size_t base = (size_t)s * H_DIM + head * HDIM + d;
  const float x1 = x[base], x2 = x[base + 32];
  const float c1 = cosT[s * HDIM + d],      s1 = sinT[s * HDIM + d];
  const float c2 = cosT[s * HDIM + d + 32], s2 = sinT[s * HDIM + d + 32];
  x[base]      = x1 * c1 - x2 * s1;
  x[base + 32] = x2 * c2 + x1 * s2;
}

// vT_bf[hd][s] = (bf16) v[s][hd]
__global__ __launch_bounds__(256) void transpose_v(const float* __restrict__ v,
                                                   __bf16* __restrict__ vT) {
  const size_t idx = (size_t)blockIdx.x * 256 + threadIdx.x;  // S*H
  const int s  = (int)(idx >> 10);
  const int hd = (int)(idx & 1023);
  vT[(size_t)hd * S_LEN + s] = (__bf16)v[idx];
}

__global__ __launch_bounds__(256) void softmax_rows(float* __restrict__ w) {
  const int row = blockIdx.x, head = blockIdx.y;
  float* p = w + ((size_t)head * S_LEN + row) * S_LEN;
  __shared__ float red[256];

  float mx = -INFINITY;
  for (int i = threadIdx.x; i < S_LEN; i += 256) mx = fmaxf(mx, p[i]);
  red[threadIdx.x] = mx; __syncthreads();
  for (int s = 128; s > 0; s >>= 1) {
    if (threadIdx.x < s) red[threadIdx.x] = fmaxf(red[threadIdx.x], red[threadIdx.x + s]);
    __syncthreads();
  }
  mx = red[0]; __syncthreads();

  float sum = 0.0f;
  for (int i = threadIdx.x; i < S_LEN; i += 256) {
    const float e = __expf(p[i] - mx);
    p[i] = e; sum += e;
  }
  red[threadIdx.x] = sum; __syncthreads();
  for (int s = 128; s > 0; s >>= 1) {
    if (threadIdx.x < s) red[threadIdx.x] += red[threadIdx.x + s];
    __syncthreads();
  }
  const float inv = 1.0f / red[0];
  for (int i = threadIdx.x; i < S_LEN; i += 256) p[i] *= inv;
}

// ------------------------------------------------------------ launcher

extern "C" void kernel_launch(void* const* d_in, const int* in_sizes, int n_in,
                              void* d_out, int out_size, void* d_ws, size_t ws_size,
                              hipStream_t stream) {
  const float* hidden = (const float*)d_in[0];
  const float* cosT   = (const float*)d_in[1];
  const float* sinT   = (const float*)d_in[2];
  // d_in[3] = attention_mask (causal; handled analytically)
  const float* W_ih = (const float*)d_in[4];
  const float* W_hh = (const float*)d_in[5];
  const float* b_ih = (const float*)d_in[6];
  const float* b_hh = (const float*)d_in[7];
  const float* Wq = (const float*)d_in[8];
  const float* bq = (const float*)d_in[9];
  const float* Wk = (const float*)d_in[10];
  const float* bk = (const float*)d_in[11];
  const float* Wv = (const float*)d_in[12];
  const float* bv = (const float*)d_in[13];
  const float* Wo = (const float*)d_in[14];

  float* out_attn = (float*)d_out;                               // [S, H]
  float* out_w    = out_attn + (size_t)S_LEN * H_DIM;            // [NH, S, S]

  char* p = (char*)d_ws;
  // fp32 scratch
  float* xproj = (float*)p; p += (size_t)S_LEN * G_DIM * 4;      // 64 MB
  float* hsb   = (float*)p; p += (size_t)S_LEN * H_DIM * 4;
  float* qb    = (float*)p; p += (size_t)S_LEN * H_DIM * 4;
  float* kb    = (float*)p; p += (size_t)S_LEN * H_DIM * 4;
  float* vb    = (float*)p; p += (size_t)S_LEN * H_DIM * 4;
  float* ctx   = (float*)p; p += (size_t)S_LEN * H_DIM * 4;
  // bf16 staging
  __bf16* hid_bf = (__bf16*)p; p += (size_t)S_LEN * H_DIM * 2;
  __bf16* Wih_bf = (__bf16*)p; p += (size_t)G_DIM * H_DIM * 2;
  __bf16* hs_bf  = (__bf16*)p; p += (size_t)S_LEN * H_DIM * 2;
  __bf16* Wq_bf  = (__bf16*)p; p += (size_t)H_DIM * H_DIM * 2;
  __bf16* Wk_bf  = (__bf16*)p; p += (size_t)H_DIM * H_DIM * 2;
  __bf16* Wv_bf  = (__bf16*)p; p += (size_t)H_DIM * H_DIM * 2;
  __bf16* Wo_bf  = (__bf16*)p; p += (size_t)H_DIM * H_DIM * 2;
  __bf16* q_bf   = (__bf16*)p; p += (size_t)S_LEN * H_DIM * 2;
  __bf16* k_bf   = (__bf16*)p; p += (size_t)S_LEN * H_DIM * 2;
  __bf16* vT_bf  = (__bf16*)p; p += (size_t)S_LEN * H_DIM * 2;
  __bf16* ctx_bf = (__bf16*)p; p += (size_t)S_LEN * H_DIM * 2;
  // barrier / lstm state
  float* hbuf  = (float*)p; p += H_DIM * 4;
  float* cbuf  = (float*)p; p += H_DIM * 4;
  unsigned* bar = (unsigned*)p; p += 64;

  hipMemsetAsync(bar, 0, 64, stream);

  const dim3 T(256);
  const int CVT_SH = (S_LEN * H_DIM) / (4 * 256);   // blocks for S*H elements
  const int CVT_HH = (H_DIM * H_DIM) / (4 * 256);   // blocks for H*H elements
  const float rsqrt_d = 0.125f;                     // 64^-0.5

  // 0) stage bf16 operands that are independent of compute order
  cvt_f32_bf16<<<dim3(CVT_SH), T, 0, stream>>>(hidden, hid_bf);
  cvt_f32_bf16<<<dim3((G_DIM * H_DIM) / 1024), T, 0, stream>>>(W_ih, Wih_bf);
  cvt_f32_bf16<<<dim3(CVT_HH), T, 0, stream>>>(Wq, Wq_bf);
  cvt_f32_bf16<<<dim3(CVT_HH), T, 0, stream>>>(Wk, Wk_bf);
  cvt_f32_bf16<<<dim3(CVT_HH), T, 0, stream>>>(Wv, Wv_bf);
  cvt_f32_bf16<<<dim3(CVT_HH), T, 0, stream>>>(Wo, Wo_bf);

  // 1) x_proj = hidden @ W_ih^T + b_ih + b_hh       M=4096 N=4096 K=1024
  gemm_wmma<<<dim3(G_DIM / 64, S_LEN / 128, 1), T, 0, stream>>>(
      hid_bf, 0LL, H_DIM, Wih_bf, 0LL, H_DIM, xproj, 0LL, G_DIM,
      b_ih, b_hh, S_LEN, G_DIM, H_DIM, 1.0f, 0);

  // 2) sequential LSTM scan -> hs (fp32), then stage bf16
  lstm_scan<<<dim3(LSTM_NB), T, 0, stream>>>(xproj, W_hh, hsb, hbuf, cbuf, bar);
  cvt_f32_bf16<<<dim3(CVT_SH), T, 0, stream>>>(hsb, hs_bf);

  // 3) q,k,v projections                            M=4096 N=1024 K=1024
  gemm_wmma<<<dim3(H_DIM / 64, S_LEN / 128, 1), T, 0, stream>>>(
      hs_bf, 0LL, H_DIM, Wq_bf, 0LL, H_DIM, qb, 0LL, H_DIM, bq, nullptr,
      S_LEN, H_DIM, H_DIM, 1.0f, 0);
  gemm_wmma<<<dim3(H_DIM / 64, S_LEN / 128, 1), T, 0, stream>>>(
      hs_bf, 0LL, H_DIM, Wk_bf, 0LL, H_DIM, kb, 0LL, H_DIM, bk, nullptr,
      S_LEN, H_DIM, H_DIM, 1.0f, 0);
  gemm_wmma<<<dim3(H_DIM / 64, S_LEN / 128, 1), T, 0, stream>>>(
      hs_bf, 0LL, H_DIM, Wv_bf, 0LL, H_DIM, vb, 0LL, H_DIM, bv, nullptr,
      S_LEN, H_DIM, H_DIM, 1.0f, 0);

  // 4) RoPE on q,k (fp32, in place), then stage bf16 q/k; transpose v -> bf16
  rope_kernel<<<dim3((S_LEN * NHEAD * 32) / 256, 2), T, 0, stream>>>(
      qb, kb, cosT, sinT);
  cvt_f32_bf16<<<dim3(CVT_SH), T, 0, stream>>>(qb, q_bf);
  cvt_f32_bf16<<<dim3(CVT_SH), T, 0, stream>>>(kb, k_bf);
  transpose_v<<<dim3((S_LEN * H_DIM) / 256), T, 0, stream>>>(vb, vT_bf);

  // 5) scores = causal(q @ k^T * scale) per head -> logits in d_out `w` region
  gemm_wmma<<<dim3(S_LEN / 64, S_LEN / 128, NHEAD), T, 0, stream>>>(
      q_bf, (long long)HDIM, H_DIM, k_bf, (long long)HDIM, H_DIM,
      out_w, (long long)S_LEN * S_LEN, S_LEN,
      nullptr, nullptr, S_LEN, S_LEN, HDIM, rsqrt_d, 1);

  // 6) softmax rows in place (produces output `w`)
  softmax_rows<<<dim3(S_LEN, NHEAD), T, 0, stream>>>(out_w);

  // 7) ctx = w @ v per head (A fp32, converted on the fly — single-use)
  gemm_wmma<<<dim3(1, S_LEN / 128, NHEAD), T, 0, stream>>>(
      out_w, (long long)S_LEN * S_LEN, S_LEN,
      vT_bf, (long long)HDIM * S_LEN, S_LEN,
      ctx, (long long)HDIM, H_DIM, nullptr, nullptr, S_LEN, HDIM, S_LEN, 1.0f, 0);
  cvt_f32_bf16<<<dim3(CVT_SH), T, 0, stream>>>(ctx, ctx_bf);

  // 8) attn_output = ctx @ Wo^T                     M=4096 N=1024 K=1024
  gemm_wmma<<<dim3(H_DIM / 64, S_LEN / 128, 1), T, 0, stream>>>(
      ctx_bf, 0LL, H_DIM, Wo_bf, 0LL, H_DIM, out_attn, 0LL, H_DIM,
      nullptr, nullptr, S_LEN, H_DIM, H_DIM, 1.0f, 0);
}
